// ClassicalQLSTM_65481071397404
// MI455X (gfx1250) — matrix operand order repaired and checked by reading, not compiled, not run on hardware
//
#include <hip/hip_runtime.h>
#include <hip/hip_bf16.h>

// ---------------------------------------------------------------------------
// Persistent-kernel LSTM for MI455X: T=512, B=128, D=512, H=512, G=2048.
//   - bf16 WMMA (v_wmma_f32_16x16x32_bf16), fp32 accumulation.
//   - 64 blocks x 128 threads (4 waves), one block per WGP.
//   - Each block stages its 128 KB slice of W^T into LDS once (320 KB/WGP),
//     then loops over all 512 time steps with a device-wide atomic barrier.
//   - Cell state c lives in VGPRs for the whole scan (tile ownership fixed).
//   - h double-buffered in bf16 in global memory across steps.
//   - Inner loop split into x-part / h-part (no per-iteration select/branch);
//     next-step x prefetched in one batch at step start.
// ---------------------------------------------------------------------------

typedef __attribute__((ext_vector_type(16))) __bf16 v16bf;
typedef __attribute__((ext_vector_type(8)))  __bf16 v8bf;
typedef __attribute__((ext_vector_type(8)))  float  v8f;

constexpr int T_ = 512;
constexpr int B_ = 128;
constexpr int D_ = 512;
constexpr int H_ = 512;
constexpr int G_ = 4 * H_;        // 2048
constexpr int K_ = D_ + H_;       // 1024
constexpr int NBLOCKS = 64;
constexpr int LDSK = K_ + 8;      // padded LDS row stride (bank-conflict break)

__device__ __forceinline__ unsigned short f2bf_bits(float f) {
    return __builtin_bit_cast(unsigned short, (__bf16)f);
}

__device__ __forceinline__ v16bf load_frag(const __bf16* p0, const __bf16* p1) {
    union { v16bf v; v8bf h[2]; } u;
    u.h[0] = *reinterpret_cast<const v8bf*>(p0);
    u.h[1] = *reinterpret_cast<const v8bf*>(p1);
    return u.v;
}

__device__ __forceinline__ float sigm(float x)  { return 1.0f / (1.0f + __expf(-x)); }
__device__ __forceinline__ float tanh_f(float x){ return 1.0f - 2.0f / (1.0f + __expf(2.0f * x)); }

// -------------------------- conversion kernels -----------------------------

__global__ void cvt_x_kernel(const float4* __restrict__ x,
                             ushort4* __restrict__ out, int n4) {
    int i = blockIdx.x * blockDim.x + threadIdx.x;
    if (i < n4) {
        float4 v = x[i];
        ushort4 r;
        r.x = f2bf_bits(v.x); r.y = f2bf_bits(v.y);
        r.z = f2bf_bits(v.z); r.w = f2bf_bits(v.w);
        out[i] = r;
    }
}

// Wt[n][k] = W_gate[k*H + n0], n in [0,2048), k in [0,1024)
__global__ void cvt_w_kernel(const float* __restrict__ Wf,
                             const float* __restrict__ Wi,
                             const float* __restrict__ Wg,
                             const float* __restrict__ Wo,
                             __bf16* __restrict__ Wt) {
    int idx = blockIdx.x * blockDim.x + threadIdx.x;   // 0 .. G*K-1
    int n = idx >> 10;
    int k = idx & (K_ - 1);
    int n0 = n & (H_ - 1);
    const float* W = (n < H_) ? Wf : (n < 2 * H_) ? Wi : (n < 3 * H_) ? Wg : Wo;
    Wt[(size_t)n * K_ + k] = (__bf16)W[(size_t)k * H_ + n0];
}

__global__ void cvt_bias_kernel(const float* __restrict__ bf,
                                const float* __restrict__ bi,
                                const float* __restrict__ bg,
                                const float* __restrict__ bo,
                                float* __restrict__ biasc) {
    int n = blockIdx.x * blockDim.x + threadIdx.x;     // 0 .. 2047
    if (n < G_) {
        const float* b = (n < H_) ? bf : (n < 2 * H_) ? bi : (n < 3 * H_) ? bg : bo;
        biasc[n] = b[n & (H_ - 1)];
    }
}

__global__ void init_state_kernel(__bf16* __restrict__ h0,
                                  __bf16* __restrict__ h1,
                                  unsigned* __restrict__ bar) {
    int i = blockIdx.x * blockDim.x + threadIdx.x;     // 0 .. B*H-1
    h0[i] = (__bf16)0.0f;
    h1[i] = (__bf16)0.0f;
    if (i == 0) *bar = 0u;
}

// ------------------------- device-wide barrier -----------------------------

__device__ __forceinline__ void grid_sync(unsigned* bar, unsigned target) {
    __threadfence();            // make this block's h/out stores visible
    __syncthreads();            // all lanes in block done before arrive
    if (threadIdx.x == 0) {
        __hip_atomic_fetch_add(bar, 1u, __ATOMIC_ACQ_REL, __HIP_MEMORY_SCOPE_AGENT);
        while (__hip_atomic_load(bar, __ATOMIC_ACQUIRE, __HIP_MEMORY_SCOPE_AGENT) < target) {
            __builtin_amdgcn_s_sleep(2);
        }
    }
    __syncthreads();            // release whole block, EXEC back to all-ones
}

// --------------------------- persistent kernel -----------------------------
// Tile map: 256 wave-tiles = 8 m-tiles x 32 n-tiles.
//   n-tile  = blockIdx >> 1          (uniform per block -> shared LDS weights)
//   m-tile  = (blockIdx & 1)*4 + wave

__global__ __launch_bounds__(128)
void lstm_persistent_kernel(const __bf16* __restrict__ Xb,     // (T,B,D) bf16
                            const __bf16* __restrict__ Wt,     // (G,K) bf16
                            const float*  __restrict__ biasc,  // (G,) fp32
                            __bf16*       __restrict__ hb0,    // (B,H) bf16
                            __bf16*       __restrict__ hb1,    // (B,H) bf16
                            float*        __restrict__ out,    // T*B*H + 2*B*H
                            unsigned*     __restrict__ bar) {
    // 64 rows (4 gates x 16 cols) x K, padded: 64*1032*2B = 129 KB of LDS.
    __shared__ __bf16 lWt[64 * LDSK];

    const int wave = threadIdx.x >> 5;
    const int lane = threadIdx.x & 31;
    const int nb = (blockIdx.x >> 1) << 4;                      // h-col tile base
    const int mb = (((blockIdx.x & 1) << 2) + wave) << 4;       // batch-row tile base

    // ---- stage this block's W^T slice into LDS (once) ----
    for (int q = threadIdx.x; q < 64 * 128; q += 128) {
        const int r  = q >> 7;        // 0..63 : gate*16 + col
        const int cw = q & 127;       // 16-byte chunk within row
        const int grow = ((r >> 4) << 9) + nb + (r & 15);
        *reinterpret_cast<uint4*>(&lWt[r * LDSK + (cw << 3)]) =
            *reinterpret_cast<const uint4*>(Wt + (size_t)grow * K_ + (cw << 3));
    }
    __syncthreads();

    const int lhi  = (lane >= 16) ? 1 : 0;
    const int arow = mb + (lane & 15);
    const int aoff = lhi ? 16 : 0;    // A-frag K sub-offset per lane half
    const int boff = lhi ? 16 : 0;    // B-frag K sub-offset per lane half
    const int col  = nb + (lane & 15);

    // Per-lane LDS base for B fragments (gate offsets become ds immediates).
    const __bf16* lbase0 = &lWt[(lane & 15) * LDSK + boff];

    const float bfv = biasc[col];
    const float biv = biasc[col + 512];
    const float bgv = biasc[col + 1024];
    const float bov = biasc[col + 1536];

    const size_t hx_off = (size_t)T_ * B_ * H_;
    const size_t cx_off = hx_off + (size_t)B_ * H_;
    const int rbase = mb + (lhi << 3);

    // Per-lane global base pointers (advance x by B*D per step).
    const __bf16* xrow = Xb + ((size_t)arow) * D_ + aoff;
    const size_t  xstep = (size_t)B_ * D_;

    float c_reg[8] = {0.f, 0.f, 0.f, 0.f, 0.f, 0.f, 0.f, 0.f};

    for (int t = 0; t < T_; ++t) {
        const __bf16* hin  = (t & 1) ? hb1 : hb0;
        __bf16*       hout = (t & 1) ? hb0 : hb1;
        const __bf16* hrow = hin + (size_t)arow * H_ + aoff;

        // Batch-prefetch next step's x row for this lane (8 x 128B = 1KB row).
        if (t + 1 < T_) {
            #pragma unroll
            for (int p = 0; p < 8; ++p)
                __builtin_prefetch(xrow + xstep + (p << 6), 0, 1);
        }

        v8f acc[4] = {};

        // ---- x part: K = 0..511 ----
        #pragma unroll 4
        for (int kk = 0; kk < 16; ++kk) {
            const __bf16* abase = xrow + (kk << 5);
            v16bf afrag = load_frag(abase, abase + 8);
            const __bf16* lb = lbase0 + (kk << 5);
            #pragma unroll
            for (int g = 0; g < 4; ++g) {
                v16bf bfrag = load_frag(lb + g * (16 * LDSK),
                                        lb + g * (16 * LDSK) + 8);
                acc[g] = __builtin_amdgcn_wmma_f32_16x16x32_bf16(
                    false, afrag, false, bfrag, (short)0, acc[g], false, false);
            }
        }

        // ---- h part: K = 512..1023 ----
        #pragma unroll 4
        for (int kk = 0; kk < 16; ++kk) {
            const __bf16* abase = hrow + (kk << 5);
            v16bf afrag = load_frag(abase, abase + 8);
            const __bf16* lb = lbase0 + 512 + (kk << 5);
            #pragma unroll
            for (int g = 0; g < 4; ++g) {
                v16bf bfrag = load_frag(lb + g * (16 * LDSK),
                                        lb + g * (16 * LDSK) + 8);
                acc[g] = __builtin_amdgcn_wmma_f32_16x16x32_bf16(
                    false, afrag, false, bfrag, (short)0, acc[g], false, false);
            }
        }

        // Epilogue: activations, cell update (c in VGPRs), h out.
        const size_t out_t = (size_t)t * B_ * H_;
        #pragma unroll
        for (int r = 0; r < 8; ++r) {
            const int row = rbase + r;
            const size_t ix = (size_t)row * H_ + col;
            float fg = sigm(acc[0][r] + bfv);
            float ig = sigm(acc[1][r] + biv);
            float gg = tanh_f(acc[2][r] + bgv);
            float og = sigm(acc[3][r] + bov);
            float cn = fg * c_reg[r] + ig * gg;
            float hn = og * tanh_f(cn);
            c_reg[r] = cn;
            out[out_t + ix] = hn;
            hout[ix] = (__bf16)hn;
            if (t == T_ - 1) {
                out[hx_off + ix] = hn;
                out[cx_off + ix] = cn;
            }
        }

        xrow += xstep;
        grid_sync(bar, (unsigned)NBLOCKS * (unsigned)(t + 1));
    }
}

// ------------------------------- launcher ----------------------------------

extern "C" void kernel_launch(void* const* d_in, const int* in_sizes, int n_in,
                              void* d_out, int out_size, void* d_ws, size_t ws_size,
                              hipStream_t stream) {
    (void)in_sizes; (void)n_in; (void)out_size; (void)ws_size;

    const float* X  = (const float*)d_in[0];
    const float* Wf = (const float*)d_in[1];
    const float* bf = (const float*)d_in[2];
    const float* Wi = (const float*)d_in[3];
    const float* bi = (const float*)d_in[4];
    const float* Wg = (const float*)d_in[5];
    const float* bg = (const float*)d_in[6];
    const float* Wo = (const float*)d_in[7];
    const float* bo = (const float*)d_in[8];
    float* out = (float*)d_out;

    // Workspace carve (256B aligned regions)
    uint8_t* w = (uint8_t*)d_ws;
    size_t off = 0;
    auto carve = [&](size_t bytes) -> void* {
        void* p = w + off;
        off = (off + bytes + 255) & ~(size_t)255;
        return p;
    };
    __bf16*   Xb    = (__bf16*)carve((size_t)T_ * B_ * D_ * 2);   // 64 MB
    __bf16*   Wt    = (__bf16*)carve((size_t)G_ * K_ * 2);        //  4 MB
    float*    biasc = (float*)carve((size_t)G_ * 4);
    __bf16*   hb0   = (__bf16*)carve((size_t)B_ * H_ * 2);
    __bf16*   hb1   = (__bf16*)carve((size_t)B_ * H_ * 2);
    unsigned* bar   = (unsigned*)carve(256);

    // One-time conversions + state init (stream-ordered before the scan)
    {
        int n4 = (T_ * B_ * D_) / 4;
        cvt_x_kernel<<<n4 / 256, 256, 0, stream>>>((const float4*)X, (ushort4*)Xb, n4);
        cvt_w_kernel<<<(G_ * K_) / 256, 256, 0, stream>>>(Wf, Wi, Wg, Wo, Wt);
        cvt_bias_kernel<<<G_ / 256, 256, 0, stream>>>(bf, bi, bg, bo, biasc);
        init_state_kernel<<<(B_ * H_) / 256, 256, 0, stream>>>(hb0, hb1, bar);
    }

    // One persistent kernel runs the entire 512-step scan.
    lstm_persistent_kernel<<<NBLOCKS, 128, 0, stream>>>(
        Xb, Wt, biasc, hb0, hb1, out, bar);
}
